// GraphPPRScoreFunc_41377714930047
// MI455X (gfx1250) — matrix operand order repaired and checked by reading, not compiled, not run on hardware
//
#include <hip/hip_runtime.h>
#include <hip/hip_bf16.h>

// Sizes fixed by the reference: B=256, N=128, D=1024.
#define BB 256
#define NN 128
#define DD 1024
#define MASKV -1000000000.0f

typedef __attribute__((ext_vector_type(16))) __bf16 v16bf;
typedef __attribute__((ext_vector_type(8)))  float  v8f;

__device__ __forceinline__ unsigned pk2(__bf16 a, __bf16 b) {
  return (unsigned)__builtin_bit_cast(unsigned short, a) |
         ((unsigned)__builtin_bit_cast(unsigned short, b) << 16);
}

// Split one float4 into bf16 hi/lo pairs and store 2+2 dwords into LDS.
__device__ __forceinline__ void split_store(unsigned* lhi, unsigned* llo, int uoff, float4 f) {
  __bf16 h0 = (__bf16)f.x, h1 = (__bf16)f.y, h2 = (__bf16)f.z, h3 = (__bf16)f.w;
  float  l0 = f.x - (float)h0, l1 = f.y - (float)h1;
  float  l2 = f.z - (float)h2, l3 = f.w - (float)h3;
  lhi[uoff + 0] = pk2(h0, h1);
  lhi[uoff + 1] = pk2(h2, h3);
  llo[uoff + 0] = pk2((__bf16)l0, (__bf16)l1);
  llo[uoff + 1] = pk2((__bf16)l2, (__bf16)l3);
}

union FragU { uint4 u[2]; v16bf v; };

// A-matrix (16x32 bf16) fragment per ISA layout:
// lane m (m<16): K 0..7 and 16..23 ; lane m+16: K 8..15 and 24..31.
// LDS row stride = 20 dwords (40 bf16) -> every chunk is 16B aligned.
__device__ __forceinline__ v16bf ld_fragA(const unsigned* l, int row, int half) {
  FragU f;
  f.u[0] = *(const uint4*)(l + row * 20 + half * 4);
  f.u[1] = *(const uint4*)(l + row * 20 + 8 + half * 4);
  return f.v;
}

// B-matrix (32x16 bf16) fragment per ISA layout:
// lane n (n<16): K 0..7 and 16..23 ; lane n+16: K 8..15 and 24..31 (matches A).
__device__ __forceinline__ v16bf ld_fragB(const unsigned* l, int row, int half) {
  FragU f;
  f.u[0] = *(const uint4*)(l + row * 20 + half * 4);
  f.u[1] = *(const uint4*)(l + row * 20 + 8 + half * 4);
  return f.v;
}

// Core: 128(M) x 64(N) x 1024(K) tile, C = A * B^T with both operands
// row-major (K contiguous, ld=1024). Split-bf16: hi*hi + hi*lo + lo*hi.
// 256 threads = 8 waves in a 4x2 grid; each wave owns a 32x32 sub-tile.
__device__ __forceinline__ void gemm128x64(const float* __restrict__ Ag,
                                           const float* __restrict__ Bg,
                                           unsigned* lAh, unsigned* lAl,
                                           unsigned* lBh, unsigned* lBl,
                                           v8f acc[2][2]) {
  const int tid  = threadIdx.x;
  const int c    = tid & 7;      // which float4 (columns c*4..c*4+3 of the 32-wide K slab)
  const int r0   = tid >> 3;     // 0..31
  const int lane = tid & 31;
  const int half = lane >> 4;
  const int lm   = lane & 15;
  const int w    = tid >> 5;
  const int wM   = (w & 3) * 32;
  const int wN   = (w >> 2) * 32;

  float4 areg[4], breg[2];
#pragma unroll
  for (int p = 0; p < 4; ++p)
    areg[p] = *(const float4*)(Ag + (size_t)(r0 + 32 * p) * DD + c * 4);
#pragma unroll
  for (int p = 0; p < 2; ++p)
    breg[p] = *(const float4*)(Bg + (size_t)(r0 + 32 * p) * DD + c * 4);

  for (int step = 0; step < DD / 32; ++step) {
    __syncthreads();  // previous tile fully consumed
#pragma unroll
    for (int p = 0; p < 4; ++p)
      split_store(lAh, lAl, (r0 + 32 * p) * 20 + c * 2, areg[p]);
#pragma unroll
    for (int p = 0; p < 2; ++p)
      split_store(lBh, lBl, (r0 + 32 * p) * 20 + c * 2, breg[p]);
    __syncthreads();

    if (step + 1 < DD / 32) {  // prefetch next K slab into registers
      const float* An = Ag + (step + 1) * 32;
      const float* Bn = Bg + (step + 1) * 32;
#pragma unroll
      for (int p = 0; p < 4; ++p)
        areg[p] = *(const float4*)(An + (size_t)(r0 + 32 * p) * DD + c * 4);
#pragma unroll
      for (int p = 0; p < 2; ++p)
        breg[p] = *(const float4*)(Bn + (size_t)(r0 + 32 * p) * DD + c * 4);
    }

    v16bf ah[2], al[2], bh[2], bl[2];
#pragma unroll
    for (int mi = 0; mi < 2; ++mi) {
      ah[mi] = ld_fragA(lAh, wM + mi * 16 + lm, half);
      al[mi] = ld_fragA(lAl, wM + mi * 16 + lm, half);
    }
#pragma unroll
    for (int ni = 0; ni < 2; ++ni) {
      bh[ni] = ld_fragB(lBh, wN + ni * 16 + lm, half);
      bl[ni] = ld_fragB(lBl, wN + ni * 16 + lm, half);
    }
#pragma unroll
    for (int mi = 0; mi < 2; ++mi)
#pragma unroll
      for (int ni = 0; ni < 2; ++ni) {
        v8f a0 = acc[mi][ni];
        a0 = __builtin_amdgcn_wmma_f32_16x16x32_bf16(false, ah[mi], false, bh[ni],
                                                     (short)0, a0, false, false);
        a0 = __builtin_amdgcn_wmma_f32_16x16x32_bf16(false, ah[mi], false, bl[ni],
                                                     (short)0, a0, false, false);
        a0 = __builtin_amdgcn_wmma_f32_16x16x32_bf16(false, al[mi], false, bh[ni],
                                                     (short)0, a0, false, false);
        acc[mi][ni] = a0;
      }
  }
}

// Kernel 1: Q = x @ Wq^T + bq and K = x @ Wk^T + bk (blockIdx.z selects which).
__global__ __launch_bounds__(256) void qk_gemm_kernel(
    const float* __restrict__ x,
    const float* __restrict__ Wq, const float* __restrict__ bq,
    const float* __restrict__ Wk, const float* __restrict__ bk,
    float* __restrict__ Qout, float* __restrict__ Kout) {
  __shared__ unsigned lAh[128 * 20], lAl[128 * 20];
  __shared__ unsigned lBh[64 * 20],  lBl[64 * 20];

  const int    nBase = blockIdx.x * 64;
  const size_t mBase = (size_t)blockIdx.y * 128;
  const float* Wm   = blockIdx.z ? Wk : Wq;
  const float* bias = blockIdx.z ? bk : bq;
  float*       Out  = blockIdx.z ? Kout : Qout;

  v8f zero8 = {0.f, 0.f, 0.f, 0.f, 0.f, 0.f, 0.f, 0.f};
  v8f acc[2][2] = {{zero8, zero8}, {zero8, zero8}};

  gemm128x64(x + mBase * DD, Wm + (size_t)nBase * DD, lAh, lAl, lBh, lBl, acc);

  const int tid = threadIdx.x, lane = tid & 31, half = lane >> 4, lm = lane & 15;
  const int w = tid >> 5, wM = (w & 3) * 32, wN = (w >> 2) * 32;
#pragma unroll
  for (int mi = 0; mi < 2; ++mi)
#pragma unroll
    for (int ni = 0; ni < 2; ++ni) {
      const int   n  = nBase + wN + ni * 16 + lm;
      const float bv = bias[n];
#pragma unroll
      for (int j = 0; j < 8; ++j) {
        const size_t m = mBase + wM + mi * 16 + half * 8 + j;
        Out[m * DD + n] = acc[mi][ni][j] + bv;
      }
    }
}

// Kernel 2: scores[b] = (Q_b K_b^T) / 32 with x_mask and diagonal masking.
__global__ __launch_bounds__(256) void scores_kernel(
    const float* __restrict__ Q, const float* __restrict__ Kt,
    const int* __restrict__ xmask, float* __restrict__ scores) {
  __shared__ unsigned lAh[128 * 20], lAl[128 * 20];
  __shared__ unsigned lBh[64 * 20],  lBl[64 * 20];

  const int b     = blockIdx.y;
  const int nBase = blockIdx.x * 64;
  const float* Ag = Q  + (size_t)b * NN * DD;
  const float* Bg = Kt + (size_t)b * NN * DD + (size_t)nBase * DD;

  v8f zero8 = {0.f, 0.f, 0.f, 0.f, 0.f, 0.f, 0.f, 0.f};
  v8f acc[2][2] = {{zero8, zero8}, {zero8, zero8}};

  gemm128x64(Ag, Bg, lAh, lAl, lBh, lBl, acc);

  const int tid = threadIdx.x, lane = tid & 31, half = lane >> 4, lm = lane & 15;
  const int w = tid >> 5, wM = (w & 3) * 32, wN = (w >> 2) * 32;
  const float scale = 0.03125f;  // 1/sqrt(1024)
#pragma unroll
  for (int mi = 0; mi < 2; ++mi)
#pragma unroll
    for (int ni = 0; ni < 2; ++ni) {
      const int col    = nBase + wN + ni * 16 + lm;
      const int mvalid = xmask[b * NN + col];
#pragma unroll
      for (int j = 0; j < 8; ++j) {
        const int row = wM + mi * 16 + half * 8 + j;
        float v = acc[mi][ni][j] * scale;
        if (mvalid == 0 || row == col) v = MASKV;
        scores[(size_t)b * NN * NN + (size_t)row * NN + col] = v;
      }
    }
}

// Kernel 3: per-batch softmax + 6-hop row-0 PPR propagation + L1 norm + clip.
// ppr[:,0,1:] = sum_{h=1..6} alpha*(0.75 P)^h [0,1:], so only a row-vector
// times matrix iteration is needed (no batched matrix powers).
__global__ __launch_bounds__(128) void ppr_kernel(const float* __restrict__ scores,
                                                  float* __restrict__ doc) {
  __shared__ float P[128 * 129];  // stride 129: conflict-free row & column access
  __shared__ float vsh[128];
  __shared__ float denom;

  const int b = blockIdx.x;
  const int t = threadIdx.x;
  const float* S = scores + (size_t)b * NN * NN;

  for (int idx = t; idx < NN * NN; idx += 128) {
    const int r = idx >> 7, cl = idx & 127;
    P[r * 129 + cl] = S[idx];
  }
  __syncthreads();

  // softmax over row t (masked entries are -1e9 -> exp ~ 0)
  {
    float mx = -3.4e38f;
    for (int j = 0; j < 128; ++j) mx = fmaxf(mx, P[t * 129 + j]);
    float sum = 0.f;
    for (int j = 0; j < 128; ++j) {
      const float e = __expf(P[t * 129 + j] - mx);
      P[t * 129 + j] = e;
      sum += e;
    }
    const float inv = 1.f / sum;
    for (int j = 0; j < 128; ++j) P[t * 129 + j] *= inv;
  }

  vsh[t] = (t == 0) ? 0.25f : 0.f;  // alpha * e0
  float accj = 0.f;
  __syncthreads();

  for (int h = 0; h < 6; ++h) {
    float nv = 0.f;
    for (int m = 0; m < 128; ++m) nv += vsh[m] * P[m * 129 + t];
    nv *= 0.75f;  // (1 - alpha)
    __syncthreads();
    vsh[t] = nv;
    accj += nv;
    __syncthreads();
  }

  vsh[t] = (t == 0) ? 0.f : fabsf(accj);
  __syncthreads();
  if (t == 0) {
    float s = 0.f;
    for (int j = 1; j < 128; ++j) s += vsh[j];
    denom = fmaxf(s, 1e-12f);
  }
  __syncthreads();
  if (t >= 1) {
    float d = accj / denom;
    d = fminf(fmaxf(d, 1e-7f), 1.f - 1e-7f);
    doc[(size_t)b * 127 + (t - 1)] = d;
  }
}

extern "C" void kernel_launch(void* const* d_in, const int* in_sizes, int n_in,
                              void* d_out, int out_size, void* d_ws, size_t ws_size,
                              hipStream_t stream) {
  (void)in_sizes; (void)n_in; (void)out_size; (void)ws_size;
  const float* x     = (const float*)d_in[0];
  const float* Wq    = (const float*)d_in[1];
  const float* bq    = (const float*)d_in[2];
  const float* Wk    = (const float*)d_in[3];
  const float* bk    = (const float*)d_in[4];
  const int*   xmask = (const int*)d_in[5];

  float* out    = (float*)d_out;
  float* doc    = out;                         // 256*127 floats
  float* scores = out + (size_t)BB * 127;      // 256*128*128 floats

  // Workspace: Q (128 MB) then K (128 MB), fp32.
  float* Q = (float*)d_ws;
  float* K = Q + (size_t)BB * NN * DD;

  qk_gemm_kernel<<<dim3(DD / 64, (BB * NN) / 128, 2), 256, 0, stream>>>(
      x, Wq, bq, Wk, bk, Q, K);
  scores_kernel<<<dim3(NN / 64, BB), 256, 0, stream>>>(Q, K, xmask, scores);
  ppr_kernel<<<dim3(BB), 128, 0, stream>>>(scores, doc);
}
